// HGATSolver_76270029242977
// MI455X (gfx1250) — matrix operand order repaired and checked by reading, not compiled, not run on hardware
//
#include <hip/hip_runtime.h>
#include <math.h>

#define HID    128
#define HEADS  4
#define CDIM   32
#define TDIMV  32
#define PPDV   8
#define FINV   4
#define SEINV  4
#define FOUTV  3
#define SEOUTV 3
#define NLAYERS 6

typedef __attribute__((ext_vector_type(16))) _Float16 v16h;
typedef __attribute__((ext_vector_type(8)))  float    v8f;

static inline unsigned cdiv(long a, long b) { return (unsigned)((a + b - 1) / b); }

// ---------------------------------------------------------------------------
// f32 -> f16 conversion helpers (activations converted once per layer,
// weights converted + transposed once per conv; GEMM then streams raw f16).
// ---------------------------------------------------------------------------
__global__ void f32_to_f16_kernel(const float* __restrict__ in, _Float16* __restrict__ out, long n) {
    long i = (long)blockIdx.x * blockDim.x + threadIdx.x;
    if (i < n) out[i] = (_Float16)in[i];
}

// Wt16[n][k] = (f16) W[k][n]   (128 x 128)
__global__ void wtrans_f16_kernel(const float* __restrict__ W, _Float16* __restrict__ Wt) {
    int i = blockIdx.x * blockDim.x + threadIdx.x;
    if (i >= 128 * 128) return;
    int k = i >> 7, n = i & 127;
    Wt[n * 128 + k] = (_Float16)W[k * 128 + n];
}

// ---------------------------------------------------------------------------
// WMMA GEMM:  Y[N x 128] = X16[N x 128] @ W + bias, W given transposed f16.
// 128 threads = 4 waves; block tile 64 rows x 128 cols.
// LDS staged with gfx1250 async Global->LDS copies (ASYNCcnt) — raw 16B/lane
// chunks, fenced with s_wait_asynccnt before the workgroup barrier.
// Fragments use contiguous k order (k = 16*(lane/16) + s) for both A 16x32 and B 32x16.
// ---------------------------------------------------------------------------
__global__ __launch_bounds__(128)
void gemm128_wmma(const _Float16* __restrict__ X16, const _Float16* __restrict__ Wt16,
                  const float* __restrict__ bias, float* __restrict__ Y, int N)
{
    __shared__ _Float16 Ws[128 * 128];   // Wt[n][k]  (32 KB)
    __shared__ _Float16 Xs[64 * 128];    // row-major (16 KB)

    const int tid  = threadIdx.x;
    const int wave = tid >> 5;
    const int lane = tid & 31;
    const int row0 = blockIdx.x * 64;

    // low 32 bits of a flat shared pointer == LDS byte offset (ISA 10.2 aperture map)
    const unsigned wbase = (unsigned)(uintptr_t)(void*)Ws;
    const unsigned xbase = (unsigned)(uintptr_t)(void*)Xs;

    // --- async copy W (32768 B = 2048 x 16B chunks, 16 iters x 128 lanes) ---
    {
        const char* wsrc = (const char*)Wt16;
#pragma unroll
        for (int it = 0; it < 16; ++it) {
            int t = tid + it * 128;
            unsigned ldso = wbase + (unsigned)t * 16u;
            const char* g = wsrc + (size_t)t * 16;
            asm volatile("global_load_async_to_lds_b128 %0, %1, off"
                         :: "v"(ldso), "v"(g) : "memory");
        }
    }
    // --- async copy X tile (64 rows x 256 B; 16 chunks/row; zero-fill tail) ---
    {
#pragma unroll
        for (int it = 0; it < 8; ++it) {
            int t  = tid + it * 128;
            int gr = row0 + (t >> 4);
            unsigned ldso = xbase + (unsigned)t * 16u;
            if (gr < N) {
                const char* g = (const char*)X16 + (size_t)gr * 256 + (size_t)(t & 15) * 16;
                asm volatile("global_load_async_to_lds_b128 %0, %1, off"
                             :: "v"(ldso), "v"(g) : "memory");
            } else {
                uint4 z = {0u, 0u, 0u, 0u};
                *(uint4*)((char*)Xs + (size_t)t * 16) = z;   // DS store, flushed by barrier
            }
        }
    }
    asm volatile("s_wait_asynccnt 0x0" ::: "memory");
    __syncthreads();

    const int mrow   = wave * 16;             // wave's 16-row slice
    const int m      = lane & 15;
    const int akbase = (lane < 16) ? 0 : 16;  // A: lanes 16-31 hold K 16-31
    const int bkbase = (lane < 16) ? 0 : 16;  // B: lanes 16-31 hold K 16-31

    v8f acc[8];
#pragma unroll
    for (int t = 0; t < 8; ++t)
#pragma unroll
        for (int i = 0; i < 8; ++i) acc[t][i] = 0.0f;

#pragma unroll
    for (int kc = 0; kc < 4; ++kc) {          // K = 128 in chunks of 32
        v16h a;
        const _Float16* arow = &Xs[(mrow + m) * 128 + kc * 32];
#pragma unroll
        for (int p = 0; p < 8; ++p) {         // VGPR pair p holds (k, k+1)
            int k = akbase + 2 * p;
            a[2 * p]     = arow[k];
            a[2 * p + 1] = arow[k + 1];
        }
#pragma unroll
        for (int t = 0; t < 8; ++t) {         // 8 N-tiles of 16 cols
            int ncol = t * 16 + (lane & 15);
            const _Float16* brow = &Ws[ncol * 128 + kc * 32 + bkbase];
            v16h b;
#pragma unroll
            for (int p = 0; p < 8; ++p) {
                b[2 * p]     = brow[2 * p];
                b[2 * p + 1] = brow[2 * p + 1];
            }
            acc[t] = __builtin_amdgcn_wmma_f32_16x16x32_f16(
                false, a, false, b, (short)0, acc[t], false, false);
        }
    }

    // C/D layout: VGPR r -> row (lane<16 ? r : 8+r), col = lane&15 within tile
    const int rbase = (lane < 16) ? 0 : 8;
    const int col0  = lane & 15;
#pragma unroll
    for (int t = 0; t < 8; ++t) {
        int gcol = t * 16 + col0;
        float bb = bias[gcol];
#pragma unroll
        for (int r = 0; r < 8; ++r) {
            int grow = row0 + mrow + rbase + r;
            if (grow < N) Y[grow * 128 + gcol] = acc[t][r] + bb;
        }
    }
}

// ---------------------------------------------------------------------------
// Elementwise / edge kernels
// ---------------------------------------------------------------------------
__global__ void fill_kernel(float* p, float v, int n) {
    int i = blockIdx.x * blockDim.x + threadIdx.x;
    if (i < n) p[i] = v;
}

__global__ void set_bias_kernel(float* out, const float* __restrict__ bias, int n128) {
    int i = blockIdx.x * blockDim.x + threadIdx.x;
    if (i < n128) out[i] = bias[i & 127];
}

__global__ void add_bias_kernel(float* out, const float* __restrict__ bias, int n128) {
    int i = blockIdx.x * blockDim.x + threadIdx.x;
    if (i < n128) out[i] += bias[i & 127];
}

// interleaved sinusoidal time encoding, one thread per (node, freq-pair)
__global__ void time_enc_kernel(const float* __restrict__ dt, const int* __restrict__ batch,
                                float* __restrict__ tenc, int N) {
    int idx = blockIdx.x * blockDim.x + threadIdx.x;
    if (idx >= N * 16) return;
    int n = idx >> 4, i = idx & 15;
    float div = __expf((float)i * (-logf(10000.0f) / (16.0f - 1e-6f)));
    float si  = dt[batch[n]] * div;
    tenc[n * 32 + 2 * i]     = __sinf(si);
    tenc[n * 32 + 2 * i + 1] = __cosf(si);
}

// out[n,o] = b[o] + [x | pos | tenc] . W[:,o]
__global__ void encoder_kernel(const float* __restrict__ x, int xdim,
                               const float* __restrict__ pos, const float* __restrict__ tenc,
                               const float* __restrict__ W, const float* __restrict__ b,
                               float* __restrict__ out, int N) {
    int idx = blockIdx.x * blockDim.x + threadIdx.x;
    if (idx >= N * 128) return;
    int n = idx >> 7, o = idx & 127;
    float acc = b[o];
    int k = 0;
    for (int j = 0; j < xdim; ++j, ++k) acc += x[n * xdim + j] * W[k * 128 + o];
    for (int j = 0; j < 2;    ++j, ++k) acc += pos[n * 2 + j]  * W[k * 128 + o];
    for (int j = 0; j < 32;   ++j, ++k) acc += tenc[n * 32 + j] * W[k * 128 + o];
    out[idx] = acc;
}

// one wave (32 lanes x 4 channels) per 128-wide row
__global__ void layernorm_kernel(const float* __restrict__ x, const float* __restrict__ g,
                                 const float* __restrict__ b, float* __restrict__ y, int N) {
    int row  = (blockIdx.x * blockDim.x + threadIdx.x) >> 5;
    int lane = threadIdx.x & 31;
    if (row >= N) return;
    const float* rp = x + (size_t)row * 128;
    float v[4], s = 0.0f;
#pragma unroll
    for (int i = 0; i < 4; ++i) { v[i] = rp[lane * 4 + i]; s += v[i]; }
    for (int off = 16; off; off >>= 1) s += __shfl_xor(s, off, 32);
    float mu = s * (1.0f / 128.0f);
    float vs = 0.0f;
#pragma unroll
    for (int i = 0; i < 4; ++i) { float d = v[i] - mu; vs += d * d; }
    for (int off = 16; off; off >>= 1) vs += __shfl_xor(vs, off, 32);
    float rstd = rsqrtf(vs * (1.0f / 128.0f) + 1e-5f);
    float* op = y + (size_t)row * 128;
#pragma unroll
    for (int i = 0; i < 4; ++i) {
        int c = lane * 4 + i;
        op[c] = (v[i] - mu) * rstd * g[c] + b[c];
    }
}

__device__ inline void atomicMaxF32(float* addr, float value) {
    // sign-split trick; mmax initialized to -inf (0xFF800000 works in both paths)
    if (value >= 0.0f) atomicMax((int*)addr, __float_as_int(value));
    else               atomicMin((unsigned int*)addr, __float_as_uint(value));
}

// GATv2 logits: a_h . leaky_relu(xl[src] + xr[dst]) + running segment max
__global__ void att_logits_kernel(const float* __restrict__ xl, const float* __restrict__ xr,
                                  const float* __restrict__ att,
                                  const int* __restrict__ src, const int* __restrict__ dst,
                                  int E, int ET, float* __restrict__ logits,
                                  float* __restrict__ mmax) {
    int idx = blockIdx.x * blockDim.x + threadIdx.x;
    if (idx >= ET * 4) return;
    int e = idx >> 2, h = idx & 3;
    int s = (e < E) ? src[e] : (e - E);
    int d = (e < E) ? dst[e] : (e - E);
    const float* a  = att + h * 32;
    const float* pl = xl + (size_t)s * 128 + h * 32;
    const float* pr = xr + (size_t)d * 128 + h * 32;
    float acc = 0.0f;
#pragma unroll 8
    for (int c = 0; c < 32; ++c) {
        float t = pl[c] + pr[c];
        t = (t > 0.0f) ? t : 0.2f * t;   // leaky_relu(0.2)
        acc += t * a[c];
    }
    logits[idx] = acc;
    atomicMaxF32(&mmax[d * 4 + h], acc);
}

// ex = exp(logit - m[dst]); denom[dst] += ex  (logits buffer rewritten as ex)
__global__ void att_exp_kernel(float* __restrict__ logits, const float* __restrict__ mmax,
                               float* __restrict__ denom, const int* __restrict__ dst,
                               int E, int ET) {
    int idx = blockIdx.x * blockDim.x + threadIdx.x;
    if (idx >= ET * 4) return;
    int e = idx >> 2, h = idx & 3;
    int d = (e < E) ? dst[e] : (e - E);
    float ex = __expf(logits[idx] - mmax[d * 4 + h]);
    logits[idx] = ex;
    atomicAdd(&denom[d * 4 + h], ex);
}

// out[dst, c] += (ex/denom) * xl[src, c]
__global__ void scatter_kernel(const float* __restrict__ xl, const float* __restrict__ ex,
                               const float* __restrict__ denom,
                               const int* __restrict__ src, const int* __restrict__ dst,
                               int E, int ET, float* __restrict__ out) {
    long idx = (long)blockIdx.x * blockDim.x + threadIdx.x;
    if (idx >= (long)ET * 128) return;
    int e = (int)(idx >> 7), c = (int)(idx & 127), h = c >> 5;
    int s = (e < E) ? src[e] : (e - E);
    int d = (e < E) ? dst[e] : (e - E);
    float dn    = denom[d * 4 + h];
    float alpha = ex[e * 4 + h] / fmaxf(dn, 1e-16f);
    atomicAdd(&out[(size_t)d * 128 + c], alpha * xl[(size_t)s * 128 + c]);
}

// h += relu(a_self*m_self + a_cross*m_cross)
__global__ void agg_residual_kernel(float* __restrict__ h, const float* __restrict__ ms,
                                    const float* __restrict__ mc,
                                    const float* __restrict__ a_self,
                                    const float* __restrict__ a_cross, long n) {
    long i = (long)blockIdx.x * blockDim.x + threadIdx.x;
    if (i >= n) return;
    float v = a_self[0] * ms[i] + a_cross[0] * mc[i];
    h[i] += fmaxf(v, 0.0f);
}

// gate = sigmoid([h0|h|phys] @ Wg + bg); fused = (1-g)*h0 + g*h
__global__ void gate_fuse_kernel(const float* __restrict__ h0, const float* __restrict__ h,
                                 const float* __restrict__ phys,
                                 const float* __restrict__ W, const float* __restrict__ b,
                                 float* __restrict__ fused, int N) {
    int idx = blockIdx.x * blockDim.x + threadIdx.x;
    if (idx >= N * 128) return;
    int n = idx >> 7, o = idx & 127;
    float acc = b[o];
    const float* h0r = h0 + (size_t)n * 128;
    const float* hr  = h  + (size_t)n * 128;
    for (int k = 0; k < 128; ++k) acc += h0r[k] * W[k * 128 + o];
    for (int k = 0; k < 128; ++k) acc += hr[k]  * W[(128 + k) * 128 + o];
    for (int k = 0; k < 8;  ++k)  acc += phys[n * 8 + k] * W[(256 + k) * 128 + o];
    float g = 1.0f / (1.0f + __expf(-acc));
    fused[idx] = (1.0f - g) * h0[idx] + g * h[idx];
}

__global__ void decoder_kernel(const float* __restrict__ h, const float* __restrict__ W,
                               const float* __restrict__ b, float* __restrict__ out,
                               int N, int OUT) {
    int idx = blockIdx.x * blockDim.x + threadIdx.x;
    if (idx >= N * OUT) return;
    int n = idx / OUT, o = idx % OUT;
    float acc = b[o];
    const float* hr = h + (size_t)n * 128;
    for (int k = 0; k < 128; ++k) acc += hr[k] * W[k * OUT + o];
    out[idx] = acc;
}

// ---------------------------------------------------------------------------
// Host orchestration
// ---------------------------------------------------------------------------
struct GatScratch {
    float *xl, *xr, *ealpha, *mmax, *denom;
    _Float16 *w16l, *w16r;
};

static void run_gat(const _Float16* xs16, int Nsrc, const _Float16* xd16, int Ndst,
                    const int* src, const int* dst, int E, int loops,
                    const float* Wl, const float* bl, const float* Wr, const float* br,
                    const float* att, const float* bias,
                    const GatScratch& sc, float* out, bool init, hipStream_t stream)
{
    wtrans_f16_kernel<<<64, 256, 0, stream>>>(Wl, sc.w16l);
    wtrans_f16_kernel<<<64, 256, 0, stream>>>(Wr, sc.w16r);
    gemm128_wmma<<<cdiv(Nsrc, 64), 128, 0, stream>>>(xs16, sc.w16l, bl, sc.xl, Nsrc);
    gemm128_wmma<<<cdiv(Ndst, 64), 128, 0, stream>>>(xd16, sc.w16r, br, sc.xr, Ndst);
    fill_kernel<<<cdiv(Ndst * 4, 256), 256, 0, stream>>>(sc.mmax, -__builtin_inff(), Ndst * 4);
    fill_kernel<<<cdiv(Ndst * 4, 256), 256, 0, stream>>>(sc.denom, 0.0f, Ndst * 4);
    if (init) set_bias_kernel<<<cdiv((long)Ndst * 128, 256), 256, 0, stream>>>(out, bias, Ndst * 128);
    else      add_bias_kernel<<<cdiv((long)Ndst * 128, 256), 256, 0, stream>>>(out, bias, Ndst * 128);
    int ET = E + loops;
    att_logits_kernel<<<cdiv((long)ET * 4, 256), 256, 0, stream>>>(
        sc.xl, sc.xr, att, src, dst, E, ET, sc.ealpha, sc.mmax);
    att_exp_kernel<<<cdiv((long)ET * 4, 256), 256, 0, stream>>>(
        sc.ealpha, sc.mmax, sc.denom, dst, E, ET);
    scatter_kernel<<<cdiv((long)ET * 128, 256), 256, 0, stream>>>(
        sc.xl, sc.ealpha, sc.denom, src, dst, E, ET, out);
}

extern "C" void kernel_launch(void* const* d_in, const int* in_sizes, int n_in,
                              void* d_out, int out_size, void* d_ws, size_t ws_size,
                              hipStream_t stream)
{
    // ---- raw inputs -------------------------------------------------------
    const float* x_fluid    = (const float*)d_in[0];
    const float* pos_fluid  = (const float*)d_in[1];
    const float* phys_fluid = (const float*)d_in[2];
    const float* x_se       = (const float*)d_in[3];
    const float* pos_se     = (const float*)d_in[4];
    const float* phys_se    = (const float*)d_in[5];
    const float* pos_sr     = (const float*)d_in[6];
    const float* dt         = (const float*)d_in[7];
    const int*   batch_f    = (const int*)d_in[8];
    const int*   batch_se   = (const int*)d_in[9];
    const int*   batch_sr   = (const int*)d_in[10];
    const int*   e_src[6] = { (const int*)d_in[11], (const int*)d_in[13], (const int*)d_in[15],
                              (const int*)d_in[17], (const int*)d_in[19], (const int*)d_in[21] };
    const int*   e_dst[6] = { (const int*)d_in[12], (const int*)d_in[14], (const int*)d_in[16],
                              (const int*)d_in[18], (const int*)d_in[20], (const int*)d_in[22] };
    const int Ef2f   = in_sizes[11], Ese2se = in_sizes[13], Ese2f = in_sizes[15];
    const int Ef2se  = in_sizes[17], Esr2f  = in_sizes[19], Esr2se = in_sizes[21];

    const int NF  = in_sizes[8];
    const int NSE = in_sizes[9];
    const int NSR = in_sizes[10];

    // ---- params (JAX pytree flatten: dict keys sorted alphabetically) ----
    // 23:dec_f.W 24:dec_f.b 25:dec_se.W 26:dec_se.b 27:enc_f.W 28:enc_f.b
    // 29:enc_se.W 30:enc_se.b 31:enc_sr.W 32:enc_sr.b 33:gate_f.W 34:gate_f.b
    // 35:gate_se.W 36:gate_se.b 37..: layers (44 leaves each)
    auto P = [&](int i) { return (const float*)d_in[i]; };
    const int LBASE = 37, LSTRIDE = 44;
    // per-layer: +0..3 agg {f_cross,f_self,se_cross,se_self}
    //            +4+ci*6: conv {Wl,Wr,att,bias,bl,br}; ci order: f2f,f2se,se2f,se2se,sr2f,sr2se
    //            +40,41 ln_f {g,b}; +42,43 ln_se {g,b}
    const int CI_F2F = 0, CI_F2SE = 1, CI_SE2F = 2, CI_SE2SE = 3, CI_SR2F = 4, CI_SR2SE = 5;

    // ---- workspace bump allocator ----------------------------------------
    char* ws = (char*)d_ws;
    size_t pos = 0;
    auto allocb = [&](size_t nbytes) {
        void* p = (void*)(ws + pos);
        pos = (pos + nbytes + 255) & ~(size_t)255;
        return p;
    };
    auto alloc  = [&](size_t nfloats) { return (float*)allocb(nfloats * sizeof(float)); };
    auto alloch = [&](size_t nhalf)   { return (_Float16*)allocb(nhalf * sizeof(_Float16)); };

    const int NMAX = (NF > NSE) ? NF : NSE;
    int ETmax = Ef2f + NF;
    if (Ese2se + NSE > ETmax) ETmax = Ese2se + NSE;
    if (Ese2f  > ETmax) ETmax = Ese2f;
    if (Ef2se  > ETmax) ETmax = Ef2se;
    if (Esr2f  > ETmax) ETmax = Esr2f;
    if (Esr2se > ETmax) ETmax = Esr2se;

    float* tencf  = alloc((size_t)NF  * 32);
    float* tencse = alloc((size_t)NSE * 32);
    float* tencsr = alloc((size_t)NSR * 32);
    float* h0f    = alloc((size_t)NF  * 128);
    float* h0se   = alloc((size_t)NSE * 128);
    float* h0sr   = alloc((size_t)NSR * 128);
    float* hf     = alloc((size_t)NF  * 128);
    float* hse    = alloc((size_t)NSE * 128);
    float* nfb    = alloc((size_t)NF  * 128);   // normalized fluid
    float* nseb   = alloc((size_t)NSE * 128);   // normalized se
    float* mfs    = alloc((size_t)NF  * 128);
    float* mfc    = alloc((size_t)NF  * 128);
    float* mses   = alloc((size_t)NSE * 128);
    float* msec   = alloc((size_t)NSE * 128);
    _Float16* nfb16  = alloch((size_t)NF  * 128);  // f16 copies for WMMA GEMM
    _Float16* nseb16 = alloch((size_t)NSE * 128);
    _Float16* h0sr16 = alloch((size_t)NSR * 128);
    GatScratch sc;
    sc.xl     = alloc((size_t)NMAX * 128);
    sc.xr     = alloc((size_t)NMAX * 128);
    sc.ealpha = alloc((size_t)ETmax * 4);
    sc.mmax   = alloc((size_t)NF * 4);
    sc.denom  = alloc((size_t)NF * 4);
    sc.w16l   = alloch(128 * 128);
    sc.w16r   = alloch(128 * 128);
    (void)ws_size; (void)n_in;

    // ---- encoders --------------------------------------------------------
    time_enc_kernel<<<cdiv((long)NF  * 16, 256), 256, 0, stream>>>(dt, batch_f,  tencf,  NF);
    time_enc_kernel<<<cdiv((long)NSE * 16, 256), 256, 0, stream>>>(dt, batch_se, tencse, NSE);
    time_enc_kernel<<<cdiv((long)NSR * 16, 256), 256, 0, stream>>>(dt, batch_sr, tencsr, NSR);
    encoder_kernel<<<cdiv((long)NF  * 128, 256), 256, 0, stream>>>(
        x_fluid, FINV,  pos_fluid, tencf,  P(27), P(28), h0f,  NF);
    encoder_kernel<<<cdiv((long)NSE * 128, 256), 256, 0, stream>>>(
        x_se,    SEINV, pos_se,    tencse, P(29), P(30), h0se, NSE);
    encoder_kernel<<<cdiv((long)NSR * 128, 256), 256, 0, stream>>>(
        nullptr, 0,     pos_sr,    tencsr, P(31), P(32), h0sr, NSR);

    hipMemcpyAsync(hf,  h0f,  (size_t)NF  * 128 * sizeof(float), hipMemcpyDeviceToDevice, stream);
    hipMemcpyAsync(hse, h0se, (size_t)NSE * 128 * sizeof(float), hipMemcpyDeviceToDevice, stream);
    // sr features are never normalized/updated: convert once
    f32_to_f16_kernel<<<cdiv((long)NSR * 128, 256), 256, 0, stream>>>(h0sr, h0sr16, (long)NSR * 128);

    // ---- GNN layers ------------------------------------------------------
    for (int L = 0; L < NLAYERS; ++L) {
        const int lb = LBASE + L * LSTRIDE;
        auto CP = [&](int ci, int which) { return P(lb + 4 + ci * 6 + which); };
        // which: 0=Wl 1=Wr 2=att 3=bias 4=bl 5=br
        layernorm_kernel<<<cdiv((long)NF  * 32, 256), 256, 0, stream>>>(hf,  P(lb + 40), P(lb + 41), nfb,  NF);
        layernorm_kernel<<<cdiv((long)NSE * 32, 256), 256, 0, stream>>>(hse, P(lb + 42), P(lb + 43), nseb, NSE);
        f32_to_f16_kernel<<<cdiv((long)NF  * 128, 256), 256, 0, stream>>>(nfb,  nfb16,  (long)NF  * 128);
        f32_to_f16_kernel<<<cdiv((long)NSE * 128, 256), 256, 0, stream>>>(nseb, nseb16, (long)NSE * 128);

        // self messages (with self-loops)
        run_gat(nfb16,  NF,  nfb16,  NF,  e_src[0], e_dst[0], Ef2f,   NF,
                CP(CI_F2F,0),  CP(CI_F2F,4),  CP(CI_F2F,1),  CP(CI_F2F,5),
                CP(CI_F2F,2),  CP(CI_F2F,3),  sc, mfs,  true,  stream);
        run_gat(nseb16, NSE, nseb16, NSE, e_src[1], e_dst[1], Ese2se, NSE,
                CP(CI_SE2SE,0),CP(CI_SE2SE,4),CP(CI_SE2SE,1),CP(CI_SE2SE,5),
                CP(CI_SE2SE,2),CP(CI_SE2SE,3),sc, mses, true,  stream);
        // cross messages into fluid: se2f + sr2f (sr uses un-normalized h0sr)
        run_gat(nseb16, NSE, nfb16,  NF,  e_src[2], e_dst[2], Ese2f,  0,
                CP(CI_SE2F,0), CP(CI_SE2F,4), CP(CI_SE2F,1), CP(CI_SE2F,5),
                CP(CI_SE2F,2), CP(CI_SE2F,3), sc, mfc,  true,  stream);
        run_gat(h0sr16, NSR, nfb16,  NF,  e_src[4], e_dst[4], Esr2f,  0,
                CP(CI_SR2F,0), CP(CI_SR2F,4), CP(CI_SR2F,1), CP(CI_SR2F,5),
                CP(CI_SR2F,2), CP(CI_SR2F,3), sc, mfc,  false, stream);
        // cross messages into se: f2se + sr2se
        run_gat(nfb16,  NF,  nseb16, NSE, e_src[3], e_dst[3], Ef2se,  0,
                CP(CI_F2SE,0), CP(CI_F2SE,4), CP(CI_F2SE,1), CP(CI_F2SE,5),
                CP(CI_F2SE,2), CP(CI_F2SE,3), sc, msec, true,  stream);
        run_gat(h0sr16, NSR, nseb16, NSE, e_src[5], e_dst[5], Esr2se, 0,
                CP(CI_SR2SE,0),CP(CI_SR2SE,4),CP(CI_SR2SE,1),CP(CI_SR2SE,5),
                CP(CI_SR2SE,2),CP(CI_SR2SE,3),sc, msec, false, stream);

        // agg order (sorted keys): +0 f_cross +1 f_self +2 se_cross +3 se_self
        agg_residual_kernel<<<cdiv((long)NF * 128, 256), 256, 0, stream>>>(
            hf,  mfs,  mfc,  P(lb + 1), P(lb + 0), (long)NF * 128);
        agg_residual_kernel<<<cdiv((long)NSE * 128, 256), 256, 0, stream>>>(
            hse, mses, msec, P(lb + 3), P(lb + 2), (long)NSE * 128);
    }

    // ---- gated fusion + decoders ----------------------------------------
    float* hff  = nfb;   // reuse
    float* hsef = nseb;
    gate_fuse_kernel<<<cdiv((long)NF  * 128, 256), 256, 0, stream>>>(
        h0f,  hf,  phys_fluid, P(33), P(34), hff,  NF);
    gate_fuse_kernel<<<cdiv((long)NSE * 128, 256), 256, 0, stream>>>(
        h0se, hse, phys_se,    P(35), P(36), hsef, NSE);

    float* out_f  = (float*)d_out;
    float* out_se = out_f + (size_t)NF * FOUTV;
    decoder_kernel<<<cdiv((long)NF  * FOUTV,  256), 256, 0, stream>>>(hff,  P(23), P(24), out_f,  NF,  FOUTV);
    decoder_kernel<<<cdiv((long)NSE * SEOUTV, 256), 256, 0, stream>>>(hsef, P(25), P(26), out_se, NSE, SEOUTV);
    (void)out_size;
}
